// Edge_to_Edge_Conv2d_88167088653042
// MI455X (gfx1250) — compile-verified, matrix operand order checked by reading, not run on hardware
//
#include <hip/hip_runtime.h>

typedef __attribute__((ext_vector_type(16))) _Float16 v16h;
typedef __attribute__((ext_vector_type(8)))  _Float16 v8h;
typedef __attribute__((ext_vector_type(8)))  float    v8f;
typedef __attribute__((ext_vector_type(4)))  float    v4f;

#define KDIM 128
#define CDIM 32
#define XP   136   // padded LDS row stride (halves): rows 16B-aligned + bank spread

// One workgroup (8 wave32s) per batch. Computes s[b, 0..31].
__global__ __launch_bounds__(256) void e2e_compute_s(
    const float* __restrict__ x,      // (B,1,K,K) fp32
    const float* __restrict__ w_col,  // (C,K)
    const float* __restrict__ b_col,  // (C)
    const float* __restrict__ w_row,  // (C,K)
    const float* __restrict__ b_row,  // (C)
    float* __restrict__ s_out)        // (B,C)
{
  __shared__ _Float16 Xs[KDIM * XP];   // X_b row-major, f16
  __shared__ _Float16 Wc[CDIM * XP];   // w_col rows, f16
  __shared__ _Float16 Wr[CDIM * XP];   // w_row rows, f16
  __shared__ float    sArr[CDIM];

  const int b   = blockIdx.x;
  const int tid = threadIdx.x;

  // Stage X_b (fp32 -> f16) into LDS.
  const float* xb = x + (size_t)b * KDIM * KDIM;
  for (int idx = tid; idx < KDIM * KDIM; idx += 256) {
    int r = idx >> 7, c = idx & 127;
    Xs[r * XP + c] = (_Float16)xb[idx];
  }
  // Stage weights (shared across batch, small).
  for (int idx = tid; idx < CDIM * KDIM; idx += 256) {
    int r = idx >> 7, c = idx & 127;
    Wc[r * XP + c] = (_Float16)w_col[idx];
    Wr[r * XP + c] = (_Float16)w_row[idx];
  }
  if (tid < CDIM) sArr[tid] = 0.0f;
  __syncthreads();

  const int wave = tid >> 5;   // 0..7
  const int lane = tid & 31;
  const int hi   = lane >> 4;  // which half of the wave (K-split in A/B frags)
  const int n    = lane & 15;  // output column within 16x16 tile
  const int m0   = wave * 16;  // this wave's 16-row slab of the k dimension
  const int m    = n;          // A-matrix: lane index = M row

  // Accumulators: fold the bias in by initializing every C element with it.
  v8f colAcc[2], rowAcc[2];
  #pragma unroll
  for (int nt = 0; nt < 2; ++nt) {
    float bc = b_col[nt * 16 + n];
    float br = b_row[nt * 16 + n];
    #pragma unroll
    for (int v = 0; v < 8; ++v) { colAcc[nt][v] = bc; rowAcc[nt][v] = br; }
  }

  #pragma unroll
  for (int kk = 0; kk < KDIM; kk += 32) {
    // A fragment for Col-GEMM: rows of X (M = k-slab row i, K = j). Contiguous.
    v16h a;
    {
      const v8h* p0 = (const v8h*)&Xs[(m0 + m) * XP + kk + hi * 8];
      const v8h* p1 = (const v8h*)&Xs[(m0 + m) * XP + kk + 16 + hi * 8];
      v8h lo = *p0, hh = *p1;
      #pragma unroll
      for (int t = 0; t < 8; ++t) { a[t] = lo[t]; a[8 + t] = hh[t]; }
    }
    // A fragment for Row-GEMM: columns of X (M = j, K = i). Strided reads.
    v16h a2;
    #pragma unroll
    for (int t = 0; t < 8; ++t) {
      a2[t]     = Xs[(kk + hi * 8 + t) * XP + m0 + m];
      a2[8 + t] = Xs[(kk + 16 + hi * 8 + t) * XP + m0 + m];
    }
    #pragma unroll
    for (int nt = 0; nt < 2; ++nt) {
      const int c = nt * 16 + n;
      v16h bc, br;
      {
        const v8h* q0 = (const v8h*)&Wc[c * XP + kk + hi * 8];
        const v8h* q1 = (const v8h*)&Wc[c * XP + kk + 16 + hi * 8];
        const v8h* r0 = (const v8h*)&Wr[c * XP + kk + hi * 8];
        const v8h* r1 = (const v8h*)&Wr[c * XP + kk + 16 + hi * 8];
        v8h u0 = *q0, u1 = *q1, w0 = *r0, w1 = *r1;
        #pragma unroll
        for (int t = 0; t < 8; ++t) {
          bc[t] = u0[t]; bc[8 + t] = u1[t];
          br[t] = w0[t]; br[8 + t] = w1[t];
        }
      }
      // D = A*B + C  (f32 accumulate)
      colAcc[nt] = __builtin_amdgcn_wmma_f32_16x16x32_f16(
          false, a,  false, bc, (short)0, colAcc[nt], false, false);
      rowAcc[nt] = __builtin_amdgcn_wmma_f32_16x16x32_f16(
          false, a2, false, br, (short)0, rowAcc[nt], false, false);
    }
  }

  // s[b,c] = sum_k Col[k,c]*Row[k,c]. Each lane owns 8 (M,N) cells of each tile:
  // N = n, M = v + 8*hi within this wave's slab. Reduce via LDS float atomics.
  #pragma unroll
  for (int nt = 0; nt < 2; ++nt) {
    float p = 0.0f;
    #pragma unroll
    for (int v = 0; v < 8; ++v) p += colAcc[nt][v] * rowAcc[nt][v];
    atomicAdd(&sArr[nt * 16 + n], p);
  }
  __syncthreads();
  if (tid < CDIM) s_out[b * CDIM + tid] = sArr[tid];
}

// Broadcast s[b,c] over a 128x128 plane (64 KB contiguous per block).
// Output (256 MB) exceeds the 192 MB L2 -> non-temporal streaming stores.
__global__ __launch_bounds__(256) void e2e_broadcast(
    const float* __restrict__ s, float* __restrict__ out)
{
  const int plane = blockIdx.x;            // b*C + c
  const float val = s[plane];
  v4f v = { val, val, val, val };
  v4f* o = (v4f*)(out + (size_t)plane * (KDIM * KDIM));
  const int t = threadIdx.x;
  #pragma unroll
  for (int r = 0; r < 16; ++r)
    __builtin_nontemporal_store(v, o + t + r * 256);
}

extern "C" void kernel_launch(void* const* d_in, const int* in_sizes, int n_in,
                              void* d_out, int out_size, void* d_ws, size_t ws_size,
                              hipStream_t stream) {
  const float* x     = (const float*)d_in[0];
  const float* w_col = (const float*)d_in[1];
  const float* b_col = (const float*)d_in[2];
  const float* w_row = (const float*)d_in[3];
  const float* b_row = (const float*)d_in[4];
  float* out = (float*)d_out;
  float* s   = (float*)d_ws;                       // B*C floats = 16 KB scratch

  const int B = in_sizes[0] / (KDIM * KDIM);       // 128
  const int C = in_sizes[2];                       // 32

  e2e_compute_s<<<B, 256, 0, stream>>>(x, w_col, b_col, w_row, b_row, s);
  e2e_broadcast<<<B * C, 256, 0, stream>>>(s, out);
  (void)n_in; (void)out_size; (void)ws_size;
}
